// DotProductAttention_8735963480093
// MI455X (gfx1250) — matrix operand order repaired and checked by reading, not compiled, not run on hardware
//
#include <hip/hip_runtime.h>
#include <hip/hip_bf16.h>

typedef __attribute__((ext_vector_type(16))) __bf16 v16bf;
typedef __attribute__((ext_vector_type(8)))  float  v8f;

// ---------------- tiling constants ----------------
static constexpr int TM  = 128;   // block tile rows (M)
static constexpr int TN  = 256;   // block tile cols (N)
static constexpr int TK  = 32;    // K per WMMA step
static constexpr int LDP = 40;    // LDS row stride (elements) -> 80B, 16B aligned, conflict-spread

// problem constants (from reference)
static constexpr int NB = 8;
static constexpr int LQ = 2048;
static constexpr int SK = 2048;
static constexpr int QD = 1024;
static constexpr int OD = 1024;

// ---------------- register staging helpers (global -> regs -> bf16 LDS) ----------------
template <typename T, int ELEMS>
__device__ __forceinline__ void stage_load(const T* __restrict__ src, float4* __restrict__ r) {
    constexpr int NV = ELEMS * (int)sizeof(T) / 16;
    #pragma unroll
    for (int i = 0; i < NV; ++i) r[i] = ((const float4*)src)[i];
}

template <typename T, int ELEMS>
__device__ __forceinline__ void stage_store(const float4* __restrict__ r, __bf16* __restrict__ dst) {
    if constexpr (sizeof(T) == 4) {
        // ELEMS f32 in regs -> ELEMS bf16 in LDS (one float4 out per 8 values)
        #pragma unroll
        for (int g = 0; g < ELEMS / 8; ++g) {
            const float* f = (const float*)&r[g * 2];
            union { float4 o; __bf16 h[8]; } u;
            #pragma unroll
            for (int e = 0; e < 8; ++e) u.h[e] = (__bf16)f[e];
            ((float4*)dst)[g] = u.o;
        }
    } else {
        #pragma unroll
        for (int g = 0; g < ELEMS * 2 / 16; ++g) ((float4*)dst)[g] = r[g];
    }
}

// ---------------- WMMA fragment loads per CDNA5 ISA VGPR layouts ----------------
// A (16x32 bf16): lane m (0-15) holds row m, K {0..7,16..23}; lane m+16 holds row m, K {8..15,24..31}
__device__ __forceinline__ v16bf ld_frag_a(const __bf16* __restrict__ base) {
    const int lane = threadIdx.x & 31;
    const int row  = lane & 15;
    const int k0   = (lane & 16) ? 8 : 0;
    const __bf16* p = base + row * LDP + k0;
    union { float4 f[2]; v16bf v; } u;
    u.f[0] = *(const float4*)(p);
    u.f[1] = *(const float4*)(p + 16);
    return u.v;
}
// B (32x16 bf16): lane n (0-15) holds column n, K {0..7,16..23}; lane n+16 holds column n, K {8..15,24..31}
__device__ __forceinline__ v16bf ld_frag_b(const __bf16* __restrict__ base) {
    const int lane = threadIdx.x & 31;
    const int col  = lane & 15;
    const int k0   = (lane & 16) ? 8 : 0;
    const __bf16* p = base + col * LDP + k0;
    union { float4 f[2]; v16bf v; } u;
    u.f[0] = *(const float4*)(p);
    u.f[1] = *(const float4*)(p + 16);
    return u.v;
}

// ---------------- unified GEMM:  C[m,n] = scale * sum_k A[m,k]*Bsrc[n,k] (+ bias[n]) ----------------
// A: row-major [M,K] (f32 or bf16), Bsrc: row-major [Ncols,K] (its rows are B's columns)
// TRANSPOSE store: C[b][n][s] with b = gm / transS, s = gm % transS  (used to emit v^T)
template <typename TA, typename TB, typename TC, bool ADD_BIAS, bool TRANSPOSE>
__global__ __launch_bounds__(256) void gemm_wmma_kernel(
    const TA* __restrict__ A, const TB* __restrict__ Bsrc, TC* __restrict__ C,
    const float* __restrict__ bias, int M, int Ncols, int K, float scale,
    long sAbatch, long sBbatch, long sCbatch, int transS)
{
    __shared__ __bf16 As[2][TM * LDP];   // 2 x 10 KB
    __shared__ __bf16 Bs[2][TN * LDP];   // 2 x 20 KB

    const int bx = blockIdx.x, by = blockIdx.y, bz = blockIdx.z;
    const TA* __restrict__ Ab = A    + (long)bz * sAbatch + (long)by * TM * K;
    const TB* __restrict__ Bb = Bsrc + (long)bz * sBbatch + (long)bx * TN * K;

    const int t    = threadIdx.x;
    const int wave = t >> 5;
    const int wm   = wave >> 2;   // 0..1  -> 64 rows per wave
    const int wn   = wave & 3;    // 0..3  -> 64 cols per wave

    // staging coordinates: A: 256 thr stage 128x32 (16 elem each); B: 256 thr stage 256x32 (32 elem each)
    const int arow = t >> 1, ahalf = (t & 1) * 16;
    const int brow = t;
    const TA* __restrict__ aSrc = Ab + (long)arow * K + ahalf;
    const TB* __restrict__ bSrc = Bb + (long)brow * K;
    __bf16* aDst0 = &As[0][arow * LDP + ahalf];
    __bf16* bDst0 = &Bs[0][brow * LDP];
    const int bufStrideA = TM * LDP, bufStrideB = TN * LDP;

    constexpr int NVA = 16 * (int)sizeof(TA) / 16;
    constexpr int NVB = 32 * (int)sizeof(TB) / 16;
    float4 ra[NVA], rb[NVB];

    v8f acc[4][4];
    #pragma unroll
    for (int i = 0; i < 4; ++i)
        #pragma unroll
        for (int j = 0; j < 4; ++j)
            #pragma unroll
            for (int e = 0; e < 8; ++e) acc[i][j][e] = 0.0f;

    const int nk = K / TK;

    // prologue: stage tile 0 into buffer 0
    stage_load<TA, 16>(aSrc, ra);
    stage_load<TB, 32>(bSrc, rb);
    stage_store<TA, 16>(ra, aDst0);
    stage_store<TB, 32>(rb, bDst0);
    __syncthreads();

    for (int kt = 0; kt < nk; ++kt) {
        const int buf = kt & 1;

        // issue global loads for the next K-tile before computing this one
        if (kt + 1 < nk) {
            stage_load<TA, 16>(aSrc + (kt + 1) * TK, ra);
            stage_load<TB, 32>(bSrc + (kt + 1) * TK, rb);
        }

        // compute on current buffer: 4x4 fragments, 16 WMMAs
        v16bf af[4], bfr[4];
        #pragma unroll
        for (int i = 0; i < 4; ++i) af[i]  = ld_frag_a(&As[buf][(wm * 64 + i * 16) * LDP]);
        #pragma unroll
        for (int j = 0; j < 4; ++j) bfr[j] = ld_frag_b(&Bs[buf][(wn * 64 + j * 16) * LDP]);
        #pragma unroll
        for (int i = 0; i < 4; ++i)
            #pragma unroll
            for (int j = 0; j < 4; ++j)
                acc[i][j] = __builtin_amdgcn_wmma_f32_16x16x32_bf16(
                    false, af[i], false, bfr[j], (short)0, acc[i][j], false, false);

        // convert + park next tile in the alternate buffer (disjoint from the one just read)
        if (kt + 1 < nk) {
            stage_store<TA, 16>(ra, aDst0 + (buf ^ 1) * bufStrideA);
            stage_store<TB, 32>(rb, bDst0 + (buf ^ 1) * bufStrideB);
        }
        __syncthreads();
    }

    // ---- store (C/D layout: lane 0-15 -> M=r, lane 16-31 -> M=r+8; N = lane&15) ----
    const int lane = t & 31;
    const int ln   = lane & 15;
    const int lm   = (lane >> 4) * 8;
    #pragma unroll
    for (int i = 0; i < 4; ++i) {
        #pragma unroll
        for (int j = 0; j < 4; ++j) {
            const int gm0 = by * TM + wm * 64 + i * 16 + lm;
            const int gn  = bx * TN + wn * 64 + j * 16 + ln;
            const float bv = ADD_BIAS ? bias[gn] : 0.0f;
            #pragma unroll
            for (int r = 0; r < 8; ++r) {
                const float v = acc[i][j][r] * scale + bv;
                const int gm = gm0 + r;
                if constexpr (TRANSPOSE) {
                    const int nb = gm / transS;
                    const int ss = gm - nb * transS;
                    C[(long)nb * Ncols * transS + (long)gn * transS + ss] = (TC)v;
                } else {
                    C[(long)bz * sCbatch + (long)gm * Ncols + gn] = (TC)v;
                }
            }
        }
    }
}

// ---------------- row softmax, in place on f32 [rows, S], S = 2048, 256 threads/row ----------------
__global__ __launch_bounds__(256) void softmax_rows(float* __restrict__ attn, int S) {
    const long row = blockIdx.x;
    float* __restrict__ p = attn + row * (long)S;
    const int t = threadIdx.x;
    __shared__ float red[256];

    float vals[8];
    float lm = -3.402823466e38f;
    #pragma unroll
    for (int i = 0; i < 8; ++i) { vals[i] = p[t + i * 256]; lm = fmaxf(lm, vals[i]); }
    red[t] = lm; __syncthreads();
    for (int off = 128; off > 0; off >>= 1) {
        if (t < off) red[t] = fmaxf(red[t], red[t + off]);
        __syncthreads();
    }
    const float m = red[0]; __syncthreads();

    float ls = 0.0f;
    #pragma unroll
    for (int i = 0; i < 8; ++i) { vals[i] = __expf(vals[i] - m); ls += vals[i]; }
    red[t] = ls; __syncthreads();
    for (int off = 128; off > 0; off >>= 1) {
        if (t < off) red[t] += red[t + off];
        __syncthreads();
    }
    const float inv = 1.0f / red[0];
    #pragma unroll
    for (int i = 0; i < 8; ++i) p[t + i * 256] = vals[i] * inv;
}

// ---------------- host side ----------------
extern "C" void kernel_launch(void* const* d_in, const int* in_sizes, int n_in,
                              void* d_out, int out_size, void* d_ws, size_t ws_size,
                              hipStream_t stream) {
    (void)in_sizes; (void)n_in; (void)out_size; (void)ws_size;
    const float* Q  = (const float*)d_in[0];  // [NB, LQ, QD]
    const float* Kx = (const float*)d_in[1];  // [NB, SK, QD]
    const float* V  = (const float*)d_in[2];  // [NB, SK, QD]
    const float* W  = (const float*)d_in[3];  // [OD, QD]
    const float* b  = (const float*)d_in[4];  // [OD]

    float* out  = (float*)d_out;                       // [NB, LQ, OD]
    float* attn = out + (long)NB * LQ * OD;            // [NB, LQ, SK]

    const long NLO = (long)NB * LQ * OD;               // == NB*SK*OD here
    __bf16* qb = (__bf16*)d_ws;                        // [NB, LQ, OD] bf16
    __bf16* kb = qb + NLO;                             // [NB, SK, OD] bf16
    __bf16* vT = kb + NLO;                             // [NB, OD, SK] bf16 (v transposed)

    const dim3 blk(256);
    const float inv_sqrt_od = 0.03125f;                // 1/sqrt(1024)

    // q = Q W^T + b  (M = NB*LQ fused)
    dim3 gp(OD / TN, (NB * LQ) / TM, 1);
    gemm_wmma_kernel<float, float, __bf16, true, false><<<gp, blk, 0, stream>>>(
        Q, W, qb, b, NB * LQ, OD, QD, 1.0f, 0, 0, 0, 0);
    // k = K W^T + b
    gemm_wmma_kernel<float, float, __bf16, true, false><<<gp, blk, 0, stream>>>(
        Kx, W, kb, b, NB * SK, OD, QD, 1.0f, 0, 0, 0, 0);
    // v = V W^T + b, stored transposed as vT[n][o][s]
    gemm_wmma_kernel<float, float, __bf16, true, true><<<gp, blk, 0, stream>>>(
        V, W, vT, b, NB * SK, OD, QD, 1.0f, 0, 0, 0, SK);

    // scores: attn_raw[n][l][s] = (1/32) * q[n][l][:] . k[n][s][:]
    dim3 gs(SK / TN, LQ / TM, NB);
    gemm_wmma_kernel<__bf16, __bf16, float, false, false><<<gs, blk, 0, stream>>>(
        qb, kb, attn, nullptr, LQ, SK, OD, inv_sqrt_od,
        (long)LQ * OD, (long)SK * OD, (long)LQ * SK, 0);

    // softmax rows (in place, f32 output kept in d_out)
    softmax_rows<<<NB * LQ, blk, 0, stream>>>(attn, SK);

    // output: out[n][l][o] = attn[n][l][:] . v[n][:][o]  (A = f32 attn, converted in LDS staging)
    dim3 go(OD / TN, LQ / TM, NB);
    gemm_wmma_kernel<float, __bf16, float, false, false><<<go, blk, 0, stream>>>(
        attn, vT, out, nullptr, LQ, OD, SK, 1.0f,
        (long)LQ * SK, (long)OD * SK, (long)LQ * OD, 0);
}
